// get_model_5145370821263
// MI455X (gfx1250) — matrix operand order repaired and checked by reading, not compiled, not run on hardware
//
#include <hip/hip_runtime.h>

// PointNet++-MSG forward for MI455X (gfx1250, wave32).
// All channel MLP/FC/VQ GEMMs go through v_wmma_f32_16x16x32_f16.
// GEMM structure: each block stages a 32-column slab of the weight matrix
// into LDS once (ds path, shared by 8 waves), each wave computes a 16x32
// output tile (NB=2 register blocking: one A fragment -> two WMMAs).
// Activations are f16 (halves HBM traffic at 23.3 TB/s, native WMMA operand).

typedef __attribute__((ext_vector_type(16))) _Float16 v16h;
typedef __attribute__((ext_vector_type(8)))  _Float16 v8h;
typedef __attribute__((ext_vector_type(8)))  float    v8f;

#define BNEPS 1e-5f

// ---------------- layout helpers ----------------
__global__ void k_split(const float* __restrict__ xyz, float* __restrict__ coords,
                        float* __restrict__ nrm, int B, int N) {
  int gid = blockIdx.x * blockDim.x + threadIdx.x;
  if (gid >= B * N) return;
  int b = gid / N, n = gid % N;
  const float* s = xyz + (size_t)b * 6 * N + n;
  float* c = coords + (size_t)gid * 3;
  float* m = nrm + (size_t)gid * 3;
  c[0] = s[0];             c[1] = s[(size_t)N];     c[2] = s[(size_t)2 * N];
  m[0] = s[(size_t)3 * N]; m[1] = s[(size_t)4 * N]; m[2] = s[(size_t)5 * N];
}

// ---------------- farthest point sampling (one block per batch) ----------------
__global__ void k_fps(const float* __restrict__ pts, int N, int S, int* __restrict__ idx) {
  int b = blockIdx.x, t = threadIdx.x;
  __shared__ float dist[1024];
  __shared__ float rd[1024];
  __shared__ int   ri[1024];
  __shared__ float cc[3];
  __shared__ int   far;
  if (t == 0) far = 0;
  dist[t] = 1e10f;
  __syncthreads();
  for (int i = 0; i < S; ++i) {
    if (t == 0) {
      idx[b * S + i] = far;
      const float* p = pts + ((size_t)b * N + far) * 3;
      cc[0] = p[0]; cc[1] = p[1]; cc[2] = p[2];
    }
    __syncthreads();
    const float* q = pts + ((size_t)b * N + t) * 3;
    float dx = q[0] - cc[0], dy = q[1] - cc[1], dz = q[2] - cc[2];
    float d = dx * dx + dy * dy + dz * dz;
    float nd = fminf(dist[t], d);
    dist[t] = nd;
    rd[t] = nd; ri[t] = t;
    __syncthreads();
    for (int s = N >> 1; s > 0; s >>= 1) {
      if (t < s) {
        if (rd[t + s] > rd[t]) { rd[t] = rd[t + s]; ri[t] = ri[t + s]; }
      }
      __syncthreads();
    }
    if (t == 0) far = ri[0];
    __syncthreads();
  }
}

__global__ void k_gather3(const float* __restrict__ pts, const int* __restrict__ idx,
                          float* __restrict__ dst, int Npts, int S, int B) {
  int gid = blockIdx.x * blockDim.x + threadIdx.x;
  if (gid >= B * S) return;
  int b = gid / S;
  int j = idx[gid];
  const float* p = pts + ((size_t)b * Npts + j) * 3;
  float* d = dst + (size_t)gid * 3;
  d[0] = p[0]; d[1] = p[1]; d[2] = p[2];
}

// ---------------- ball query (one thread per centroid) ----------------
__global__ void k_ball(const float* __restrict__ pts, const float* __restrict__ nxyz,
                       int* __restrict__ gidx, int N, int S, int K, float r2, int B) {
  int gid = blockIdx.x * blockDim.x + threadIdx.x;
  if (gid >= B * S) return;
  int b = gid / S;
  const float* c = nxyz + (size_t)gid * 3;
  int* out = gidx + (size_t)gid * K;
  int cnt = 0, first = -1;
  for (int n = 0; n < N && cnt < K; ++n) {
    const float* p = pts + ((size_t)b * N + n) * 3;
    float dx = p[0] - c[0], dy = p[1] - c[1], dz = p[2] - c[2];
    float d = dx * dx + dy * dy + dz * dz;
    if (d <= r2) { if (first < 0) first = n; out[cnt++] = n; }
  }
  if (first < 0) first = 0;
  for (; cnt < K; ++cnt) out[cnt] = first;
}

// ---------------- grouping: [points(gidx), xyz(gidx)-new_xyz] -> padded f16 rows ----
__global__ void k_group(const float* __restrict__ pts, const float* __restrict__ feats,
                        int Cf, const float* __restrict__ nxyz, const int* __restrict__ gidx,
                        _Float16* __restrict__ A, int Npts, int S, int K, int Cp, int B) {
  int gid = blockIdx.x * blockDim.x + threadIdx.x;
  if (gid >= B * S * K) return;
  int b = gid / (S * K);
  int r = gid % (S * K);
  int s = r / K;
  int n = gidx[gid];
  const float* cen = nxyz + ((size_t)b * S + s) * 3;
  _Float16* o = A + (size_t)gid * Cp;
  int c = 0;
  const float* f = feats + ((size_t)b * Npts + n) * Cf;
  for (; c < Cf; ++c) o[c] = (_Float16)f[c];
  const float* p = pts + ((size_t)b * Npts + n) * 3;
  o[c++] = (_Float16)(p[0] - cen[0]);
  o[c++] = (_Float16)(p[1] - cen[1]);
  o[c++] = (_Float16)(p[2] - cen[2]);
  for (; c < Cp; ++c) o[c] = (_Float16)0.f;
}

// sa3 group-all: [l2_xyz, l2_p] -> (B*128, 672) f16
__global__ void k_groupall(const float* __restrict__ l2xyz, const float* __restrict__ l2p,
                           _Float16* __restrict__ A) {
  int gid = blockIdx.x * blockDim.x + threadIdx.x;   // B*128
  if (gid >= 8 * 128) return;
  _Float16* o = A + (size_t)gid * 672;
  const float* x = l2xyz + (size_t)gid * 3;
  o[0] = (_Float16)x[0]; o[1] = (_Float16)x[1]; o[2] = (_Float16)x[2];
  const float* f = l2p + (size_t)gid * 640;
  for (int c = 0; c < 640; ++c) o[3 + c] = (_Float16)f[c];
  for (int c = 643; c < 672; ++c) o[c] = (_Float16)0.f;
}

// ---------------- weight f32 -> padded f16 ----------------
__global__ void k_convw(const float* __restrict__ Ws, _Float16* __restrict__ Wh,
                        int n, int k, int np, int kp) {
  int gid = blockIdx.x * blockDim.x + threadIdx.x;
  if (gid >= np * kp) return;
  int r = gid / kp, c = gid % kp;
  Wh[gid] = (r < n && c < k) ? (_Float16)Ws[(size_t)r * k + c] : (_Float16)0.f;
}

// ---------------- WMMA GEMM: Y[M,N] = A[M,K] * W[N,K]^T (f16 in, f32 acc) -------
// K % 32 == 0; M % 16 == 0; N % 32 == 0. Block = 8 waves; a block claims a
// 32-column W slab, stages it into LDS once, each wave computes a 16x32 tile.
__global__ void k_gemm_wmma(const _Float16* __restrict__ A, const _Float16* __restrict__ W,
                            _Float16* __restrict__ Yh, float* __restrict__ Yf,
                            int M, int N, int K) {
  __shared__ __align__(16) _Float16 sB[32 * 1024];   // 64 KB: 32 W rows x K<=1024
  const int lane   = threadIdx.x & 31;
  const int half   = lane >> 4;     // K-half selector (wave32 WMMA layout)
  const int l15    = lane & 15;
  const int waveId = threadIdx.x >> 5;
  const int tilesM = M >> 4;
  const int tilesP = N >> 5;                 // 32-wide column slabs
  const int tmG    = (tilesM + 7) >> 3;      // 8 m-tiles per block
  const long blockTiles = (long)tmG * tilesP;
  const int kv = K >> 3;                     // v8h chunks per W row
  for (long bt = blockIdx.x; bt < blockTiles; bt += gridDim.x) {
    const int tmg = (int)(bt / tilesP);
    const int tp  = (int)(bt % tilesP);
    __syncthreads();                         // protect sB from previous iter readers
    for (int idx = threadIdx.x; idx < 32 * kv; idx += blockDim.x) {
      int row = idx / kv, c8 = idx % kv;
      *(v8h*)&sB[row * K + c8 * 8] =
          *(const v8h*)&W[((size_t)(tp * 32 + row)) * K + c8 * 8];
    }
    __syncthreads();
    const int tm = tmg * 8 + waveId;         // wave-uniform guard: EXEC stays full
    if (tm < tilesM) {
      // A: lane row = l15; element e<8 -> k = half*8 + e, e>=8 -> k = 16+half*8+(e-8)
      const _Float16* arow = A + ((size_t)(tm * 16 + l15)) * K + half * 8;
      // B (from LDS): lane col = l15; element e<8 -> k = half*8 + e, e>=8 -> k = 16+half*8+(e-8)
      const _Float16* b0 = sB + (size_t)l15 * K + half * 8;
      const _Float16* b1 = b0 + (size_t)16 * K;
      v8f acc0 = {}, acc1 = {};
      for (int kc = 0; kc < K; kc += 32) {
        v16h a, f0, f1;
        *(v8h*)&a        = *(const v8h*)(arow + kc);
        *((v8h*)&a + 1)  = *(const v8h*)(arow + kc + 16);
        *(v8h*)&f0       = *(const v8h*)(b0 + kc);
        *((v8h*)&f0 + 1) = *(const v8h*)(b0 + kc + 16);
        *(v8h*)&f1       = *(const v8h*)(b1 + kc);
        *((v8h*)&f1 + 1) = *(const v8h*)(b1 + kc + 16);
        __builtin_prefetch(arow + kc + 32, 0, 1);   // global_prefetch_b8
        acc0 = __builtin_amdgcn_wmma_f32_16x16x32_f16(
            false, a, false, f0, (short)0, acc0, false, false);
        acc1 = __builtin_amdgcn_wmma_f32_16x16x32_f16(
            false, a, false, f1, (short)0, acc1, false, false);
      }
      const int col0 = tp * 32 + l15;
      #pragma unroll
      for (int r = 0; r < 8; ++r) {
        const size_t row = (size_t)(tm * 16 + half * 8 + r);
        if (Yh) {
          Yh[row * N + col0]      = (_Float16)acc0[r];
          Yh[row * N + col0 + 16] = (_Float16)acc1[r];
        }
        if (Yf) {
          Yf[row * N + col0]      = acc0[r];
          Yf[row * N + col0 + 16] = acc1[r];
        }
      }
    }
  }
}

// ---------------- batchnorm over rows, per channel ----------------
__global__ void k_bnstats(const _Float16* __restrict__ Y, float* __restrict__ mv,
                          long Mrows, int C) {
  int c = blockIdx.x, t = threadIdx.x;
  __shared__ float ss[256], sq[256];
  float s = 0.f, q = 0.f;
  for (long m = t; m < Mrows; m += blockDim.x) {
    float v = (float)Y[(size_t)m * C + c];
    s += v; q += v * v;
  }
  ss[t] = s; sq[t] = q;
  __syncthreads();
  for (int k = 128; k > 0; k >>= 1) {
    if (t < k) { ss[t] += ss[t + k]; sq[t] += sq[t + k]; }
    __syncthreads();
  }
  if (t == 0) {
    float m = ss[0] / (float)Mrows;
    mv[c] = m;
    mv[C + c] = sq[0] / (float)Mrows - m * m;
  }
}

__global__ void k_bnrelu(_Float16* __restrict__ Y, const float* __restrict__ mv,
                         long Mrows, int C) {
  size_t gid = (size_t)blockIdx.x * blockDim.x + threadIdx.x;
  if (gid >= (size_t)Mrows * C) return;
  int c = (int)(gid % C);
  float v = (float)Y[gid];
  v = (v - mv[c]) * rsqrtf(mv[C + c] + BNEPS);
  Y[gid] = (_Float16)fmaxf(v, 0.f);
}

// ---------------- max pool over K group dim ----------------
__global__ void k_maxpool(const _Float16* __restrict__ act, int BS, int K, int C,
                          float* __restrict__ dst, int Cdst, int coff) {
  int gid = blockIdx.x * blockDim.x + threadIdx.x;
  if (gid >= BS * C) return;
  int bs = gid / C, c = gid % C;
  float m = -1e30f;
  const _Float16* p = act + ((size_t)bs * K) * C + c;
  for (int k = 0; k < K; ++k) m = fmaxf(m, (float)p[(size_t)k * C]);
  dst[(size_t)bs * Cdst + coff + c] = m;
}

// ---------------- VQ ----------------
__global__ void k_buildZ(const float* __restrict__ l3p, _Float16* __restrict__ Zh,
                         float* __restrict__ loss) {
  int gid = blockIdx.x * blockDim.x + threadIdx.x;   // 16*512
  if (gid == 0) *loss = 0.f;
  if (gid >= 16 * 512) return;
  int r = gid / 512, c = gid % 512;
  int b = r & 7, off = (r < 8) ? 0 : 512;
  Zh[gid] = (_Float16)l3p[(size_t)b * 1024 + off + c];
}

__global__ void k_cnorm(const float* __restrict__ cb, float* __restrict__ cn) {
  int j = blockIdx.x * blockDim.x + threadIdx.x;
  if (j >= 1024) return;
  const float* row = cb + (size_t)j * 512;
  float s = 0.f;
  for (int c = 0; c < 512; ++c) s += row[c] * row[c];
  cn[j] = s;
}

__global__ void k_vq(const float* __restrict__ scores, const float* __restrict__ cn,
                     const float* __restrict__ cb, const float* __restrict__ l3p,
                     float* __restrict__ zq, float* __restrict__ loss) {
  int r = blockIdx.x, t = threadIdx.x;     // 16 blocks x 256
  int b = r & 7, off = (r < 8) ? 0 : 512;
  __shared__ float sd[256];
  __shared__ int   si[256];
  float best = 1e30f; int bi = 0;
  for (int j = t; j < 1024; j += 256) {
    float d = cn[j] - 2.f * scores[(size_t)r * 1024 + j];
    if (d < best) { best = d; bi = j; }
  }
  sd[t] = best; si[t] = bi;
  __syncthreads();
  for (int k = 128; k > 0; k >>= 1) {
    if (t < k) {
      if (sd[t + k] < sd[t] || (sd[t + k] == sd[t] && si[t + k] < si[t])) {
        sd[t] = sd[t + k]; si[t] = si[t + k];
      }
    }
    __syncthreads();
  }
  int j = si[0];
  __syncthreads();
  const float* q = cb + (size_t)j * 512;
  const float* z = l3p + (size_t)b * 1024 + off;
  float s = 0.f;
  for (int c = t; c < 512; c += 256) {
    float d = q[c] - z[c];
    s += d * d;
    zq[(size_t)b * 1024 + off + c] = q[c];
  }
  sd[t] = s;
  __syncthreads();
  for (int k = 128; k > 0; k >>= 1) {
    if (t < k) sd[t] += sd[t + k];
    __syncthreads();
  }
  if (t == 0) atomicAdd(loss, 0.1f * sd[0] / 4096.f);
}

__global__ void k_zqh(const float* __restrict__ zq, _Float16* __restrict__ zqh) {
  int gid = blockIdx.x * blockDim.x + threadIdx.x;   // 16*1024
  if (gid >= 16 * 1024) return;
  int r = gid / 1024, c = gid % 1024;
  zqh[gid] = (r < 8) ? (_Float16)zq[(size_t)r * 1024 + c] : (_Float16)0.f;
}

// ---------------- FC: bias + BN over batch (8 rows) + relu ----------------
__global__ void k_fcbn(const float* __restrict__ Y, const float* __restrict__ bias,
                       float* __restrict__ Xf, _Float16* __restrict__ Xh, int C) {
  int c = blockIdx.x * blockDim.x + threadIdx.x;
  if (c >= C) return;
  float v[8], s = 0.f, q = 0.f;
  for (int r = 0; r < 8; ++r) {
    float x = Y[(size_t)r * C + c] + bias[c];
    v[r] = x; s += x; q += x * x;
  }
  float m = s * 0.125f;
  float inv = rsqrtf(q * 0.125f - m * m + BNEPS);
  for (int r = 0; r < 8; ++r) {
    float x = fmaxf((v[r] - m) * inv, 0.f);
    Xf[(size_t)r * C + c] = x;
    Xh[(size_t)r * C + c] = (_Float16)x;
  }
  for (int r = 8; r < 16; ++r) Xh[(size_t)r * C + c] = (_Float16)0.f;
}

// ---------------- head: bias + log_softmax over 40 logits (y3 stride 64) --------
__global__ void k_head(const float* __restrict__ y3, const float* __restrict__ bias,
                       float* __restrict__ out) {
  int b = blockIdx.x, t = threadIdx.x;   // 8 blocks x 64
  __shared__ float sh[64];
  float x = (t < 40) ? y3[(size_t)b * 64 + t] + bias[t] : -1e30f;
  sh[t] = x;
  __syncthreads();
  for (int k = 32; k > 0; k >>= 1) { if (t < k) sh[t] = fmaxf(sh[t], sh[t + k]); __syncthreads(); }
  float mx = sh[0];
  __syncthreads();
  sh[t] = (t < 40) ? __expf(x - mx) : 0.f;
  __syncthreads();
  for (int k = 32; k > 0; k >>= 1) { if (t < k) sh[t] += sh[t + k]; __syncthreads(); }
  float lse = mx + __logf(sh[0]);
  if (t < 40) out[(size_t)b * 40 + t] = x - lse;
}

// =======================================================================
extern "C" void kernel_launch(void* const* d_in, const int* in_sizes, int n_in,
                              void* d_out, int out_size, void* d_ws, size_t ws_size,
                              hipStream_t stream) {
  (void)in_sizes; (void)n_in; (void)out_size; (void)ws_size;
  const int B = 8, N = 1024;
  const float* xyz = (const float*)d_in[0];
  const float* W1[3][3], *W2[3][3], *W3[3];
  for (int br = 0; br < 3; ++br)
    for (int l = 0; l < 3; ++l) {
      W1[br][l] = (const float*)d_in[1 + br * 3 + l];
      W2[br][l] = (const float*)d_in[10 + br * 3 + l];
    }
  for (int l = 0; l < 3; ++l) W3[l] = (const float*)d_in[19 + l];
  const float* fc1w = (const float*)d_in[22]; const float* fc1b = (const float*)d_in[23];
  const float* fc2w = (const float*)d_in[24]; const float* fc2b = (const float*)d_in[25];
  const float* fc3w = (const float*)d_in[26]; const float* fc3b = (const float*)d_in[27];
  const float* cb   = (const float*)d_in[28];

  float* out = (float*)d_out;
  float* l3p = out + 320;            // (8,1024)
  float* loss = out + 320 + 8192;

  char* ws = (char*)d_ws; size_t cur = 0;
  auto alloc = [&](size_t bytes) -> void* {
    void* p = ws + cur; cur += (bytes + 255) & ~(size_t)255; return p;
  };
  float* coords = (float*)alloc((size_t)B * N * 3 * 4);
  float* nrm    = (float*)alloc((size_t)B * N * 3 * 4);
  int*   fpsidx = (int*)  alloc((size_t)B * 512 * 4);
  float* l1xyz  = (float*)alloc((size_t)B * 512 * 3 * 4);
  float* l1p    = (float*)alloc((size_t)B * 512 * 320 * 4);
  float* l2xyz  = (float*)alloc((size_t)B * 128 * 3 * 4);
  float* l2p    = (float*)alloc((size_t)B * 128 * 640 * 4);
  int*   gidx   = (int*)  alloc((size_t)B * 512 * 128 * 4);
  _Float16* bufX = (_Float16*)alloc((size_t)134217728);
  _Float16* bufY = (_Float16*)alloc((size_t)134217728);
  _Float16* Wh   = (_Float16*)alloc((size_t)2 * 1024 * 1024);
  float* mv     = (float*)alloc(2 * 1024 * 4);
  _Float16* Zh  = (_Float16*)alloc((size_t)16 * 512 * 2);
  float* scores = (float*)alloc((size_t)16 * 1024 * 4);
  float* cn     = (float*)alloc(1024 * 4);
  float* zq     = (float*)alloc((size_t)8 * 1024 * 4);
  _Float16* zqh = (_Float16*)alloc((size_t)16 * 1024 * 2);
  float* y1     = (float*)alloc((size_t)16 * 512 * 4);
  _Float16* x1h = (_Float16*)alloc((size_t)16 * 512 * 2);
  float* y2     = (float*)alloc((size_t)16 * 256 * 4);
  _Float16* x2h = (_Float16*)alloc((size_t)16 * 256 * 2);
  float* y3     = (float*)alloc((size_t)16 * 64 * 4);

  auto pad32 = [](int c) { return (c + 31) & ~31; };
  auto conv_w = [&](const float* Wsrc, int n, int k, int np, int kp) {
    int tot = np * kp;
    k_convw<<<(tot + 255) / 256, 256, 0, stream>>>(Wsrc, Wh, n, k, np, kp);
  };
  auto gemm = [&](const _Float16* A, _Float16* Yh, float* Yf, int M, int Nn, int Kk) {
    long blockTiles = (long)(((M / 16) + 7) / 8) * (Nn / 32);
    long blocks = blockTiles;
    if (blocks < 1) blocks = 1;
    if (blocks > 4096) blocks = 4096;
    k_gemm_wmma<<<(int)blocks, 256, 0, stream>>>(A, Wh, Yh, Yf, M, Nn, Kk);
  };
  auto bn_relu = [&](_Float16* Yh, long Mrows, int C) {
    k_bnstats<<<C, 256, 0, stream>>>(Yh, mv, Mrows, C);
    size_t tot = (size_t)Mrows * C;
    k_bnrelu<<<(int)((tot + 255) / 256), 256, 0, stream>>>(Yh, mv, Mrows, C);
  };

  // ---------- level 0: split channels ----------
  k_split<<<(B * N + 255) / 256, 256, 0, stream>>>(xyz, coords, nrm, B, N);

  struct Br { int K; float r2; int cin[3]; int cout[3]; };
  const Br br1[3] = {
    {16,  0.01f, {6, 32, 32},  {32, 32, 64}},
    {32,  0.04f, {6, 64, 64},  {64, 64, 128}},
    {128, 0.16f, {6, 64, 96},  {64, 96, 128}},
  };
  const Br br2[3] = {
    {32,  0.04f, {323, 64, 64},   {64, 64, 128}},
    {64,  0.16f, {323, 128, 128}, {128, 128, 256}},
    {128, 0.64f, {323, 128, 128}, {128, 128, 256}},
  };
  const int coff1[3] = {0, 64, 192};
  const int coff2[3] = {0, 128, 384};

  // ---------- SA1 (N=1024 -> S=512) ----------
  k_fps<<<B, 1024, 0, stream>>>(coords, 1024, 512, fpsidx);
  k_gather3<<<(B * 512 + 255) / 256, 256, 0, stream>>>(coords, fpsidx, l1xyz, 1024, 512, B);
  for (int i = 0; i < 3; ++i) {
    const Br& br = br1[i];
    const int S = 512, rows = B * S * br.K;
    k_ball<<<(B * S + 255) / 256, 256, 0, stream>>>(coords, l1xyz, gidx, 1024, S, br.K, br.r2, B);
    int cp = pad32(br.cin[0]);
    k_group<<<(rows + 255) / 256, 256, 0, stream>>>(coords, nrm, 3, l1xyz, gidx, bufX,
                                                    1024, S, br.K, cp, B);
    _Float16* a = bufX; _Float16* o = bufY;
    for (int l = 0; l < 3; ++l) {
      int kp = pad32(br.cin[l]);
      conv_w(W1[i][l], br.cout[l], br.cin[l], br.cout[l], kp);
      gemm(a, o, nullptr, rows, br.cout[l], kp);
      bn_relu(o, rows, br.cout[l]);
      _Float16* t = a; a = o; o = t;
    }
    k_maxpool<<<(B * S * br.cout[2] + 255) / 256, 256, 0, stream>>>(
        a, B * S, br.K, br.cout[2], l1p, 320, coff1[i]);
  }

  // ---------- SA2 (N=512 -> S=128) ----------
  k_fps<<<B, 512, 0, stream>>>(l1xyz, 512, 128, fpsidx);
  k_gather3<<<(B * 128 + 255) / 256, 256, 0, stream>>>(l1xyz, fpsidx, l2xyz, 512, 128, B);
  for (int i = 0; i < 3; ++i) {
    const Br& br = br2[i];
    const int S = 128, rows = B * S * br.K;
    k_ball<<<(B * S + 255) / 256, 256, 0, stream>>>(l1xyz, l2xyz, gidx, 512, S, br.K, br.r2, B);
    int cp = pad32(br.cin[0]);
    k_group<<<(rows + 255) / 256, 256, 0, stream>>>(l1xyz, l1p, 320, l2xyz, gidx, bufX,
                                                    512, S, br.K, cp, B);
    _Float16* a = bufX; _Float16* o = bufY;
    for (int l = 0; l < 3; ++l) {
      int kp = pad32(br.cin[l]);
      conv_w(W2[i][l], br.cout[l], br.cin[l], br.cout[l], kp);
      gemm(a, o, nullptr, rows, br.cout[l], kp);
      bn_relu(o, rows, br.cout[l]);
      _Float16* t = a; a = o; o = t;
    }
    k_maxpool<<<(B * S * br.cout[2] + 255) / 256, 256, 0, stream>>>(
        a, B * S, br.K, br.cout[2], l2p, 640, coff2[i]);
  }

  // ---------- SA3 (group all, 1024 rows) ----------
  {
    const int rows = B * 128;
    k_groupall<<<(rows + 255) / 256, 256, 0, stream>>>(l2xyz, l2p, bufX);
    const int cin[3] = {643, 256, 512};
    const int cout[3] = {256, 512, 1024};
    _Float16* a = bufX; _Float16* o = bufY;
    for (int l = 0; l < 3; ++l) {
      int kp = pad32(cin[l]);
      conv_w(W3[l], cout[l], cin[l], cout[l], kp);
      gemm(a, o, nullptr, rows, cout[l], kp);
      bn_relu(o, rows, cout[l]);
      _Float16* t = a; a = o; o = t;
    }
    k_maxpool<<<(B * 1024 + 255) / 256, 256, 0, stream>>>(a, B, 128, 1024, l3p, 1024, 0);
  }

  // ---------- VQ ----------
  k_buildZ<<<(16 * 512 + 255) / 256, 256, 0, stream>>>(l3p, Zh, loss);
  conv_w(cb, 1024, 512, 1024, 512);
  gemm(Zh, nullptr, scores, 16, 1024, 512);
  k_cnorm<<<4, 256, 0, stream>>>(cb, cn);
  k_vq<<<16, 256, 0, stream>>>(scores, cn, cb, l3p, zq, loss);
  k_zqh<<<(16 * 1024 + 255) / 256, 256, 0, stream>>>(zq, zqh);

  // ---------- FC head ----------
  conv_w(fc1w, 512, 1024, 512, 1024);
  gemm(zqh, nullptr, y1, 16, 512, 1024);
  k_fcbn<<<2, 256, 0, stream>>>(y1, fc1b, y1 /*Xf unused later*/, x1h, 512);
  conv_w(fc2w, 256, 512, 256, 512);
  gemm(x1h, nullptr, y2, 16, 256, 512);
  k_fcbn<<<1, 256, 0, stream>>>(y2, fc2b, y2, x2h, 256);
  conv_w(fc3w, 40, 256, 64, 256);
  gemm(x2h, nullptr, y3, 16, 64, 256);
  k_head<<<8, 64, 0, stream>>>(y3, fc3b, out);
}
